// gcn_68616397521581
// MI455X (gfx1250) — matrix operand order repaired and checked by reading, not compiled, not run on hardware
//
#include <hip/hip_runtime.h>
#include <hip/hip_bf16.h>

// ---- problem constants (match reference) ----
#define Bn   64
#define Cn   32
#define Vn   1024
#define Ln   12
#define LP   16          // L padded to 16 for WMMA N
#define COn  64
#define CCAT 224
#define MBt  4           // M-tiles (16 rows each) per wave in prop_wmma
#define NBt  2           // bc (batch) tiles per block in prop_wmma

typedef __attribute__((ext_vector_type(16))) _Float16 v16h;
typedef __attribute__((ext_vector_type(8)))  _Float16 v8h;
typedef __attribute__((ext_vector_type(8)))  float    v8f;

__device__ __forceinline__ v16h ld_frag(const _Float16* p0, const _Float16* p1) {
  v8h lo = *(const v8h*)p0;
  v8h hi = *(const v8h*)p1;
  return __builtin_shufflevector(lo, hi,
                                 0, 1, 2, 3, 4, 5, 6, 7,
                                 8, 9, 10, 11, 12, 13, 14, 15);
}

// ---------------------------------------------------------------------------
// Aᵀ build: AT[w][v] = A[v][w], f32 -> f16, LDS-tiled for coalescing.
// grid.x = (V/32)*(V/32) = 1024, block = 256 (32x8 logical)
// ---------------------------------------------------------------------------
__global__ __launch_bounds__(256) void transpose_a(const float* __restrict__ A,
                                                   _Float16* __restrict__ AT) {
  __shared__ float t[32][33];
  const int tw = blockIdx.x & 31, tv = blockIdx.x >> 5;
  const int v0 = tv * 32, w0 = tw * 32;
  const int tx = threadIdx.x & 31, ty = threadIdx.x >> 5;
#pragma unroll
  for (int i = 0; i < 4; ++i)
    t[ty + 8 * i][tx] = A[(size_t)(v0 + ty + 8 * i) * Vn + w0 + tx];
  __syncthreads();
#pragma unroll
  for (int i = 0; i < 4; ++i)
    AT[(size_t)(w0 + ty + 8 * i) * Vn + v0 + tx] = (_Float16)t[tx][ty + 8 * i];
}

// ---------------------------------------------------------------------------
// Xᵀ build: XT[bc][l][v] = x[bc][v][l], f32 -> f16, l in [0,16) padded w/ 0.
// grid = (B*C, V/64), block = 256
// ---------------------------------------------------------------------------
__global__ __launch_bounds__(256) void pack_x(const float* __restrict__ x,
                                              _Float16* __restrict__ XT) {
  __shared__ float t[64 * Ln];
  const int bc = blockIdx.x;
  const int v0 = blockIdx.y * 64;
  const float* src = x + ((size_t)bc * Vn + v0) * Ln;
  for (int i = threadIdx.x; i < 64 * Ln; i += 256) t[i] = src[i];
  __syncthreads();
  _Float16* dst = XT + (size_t)bc * LP * Vn;
  for (int o = threadIdx.x; o < 64 * LP; o += 256) {
    const int l = o >> 6, v = o & 63;
    const float val = (l < Ln) ? t[v * Ln + l] : 0.0f;
    dst[(size_t)l * Vn + v0 + v] = (_Float16)val;
  }
}

// ---------------------------------------------------------------------------
// Diffusion propagation as WMMA GEMM:
//   OUT[w][l] = sum_v AT[w][v] * XT[l][v]     (per bc)
// Block stages NBt bc X-tiles in LDS (async copies); each wave keeps
// MBt x NBt accumulator tiles (A fragment reused NBt times, B fragment MBt
// times => 0.5 KB L2 + 0.25 KB LDS per WMMA). K loop is ping-pong double
// buffered with FIXED buffer names (no rotation copies): while phase P0's
// WMMAs run, phase P1's loads are in flight, and vice versa.
// grid = (B*C/NBt, 64/(8*MBt)), block = 256 (8 waves).
// ---------------------------------------------------------------------------
__global__ __launch_bounds__(256) void prop_wmma(const _Float16* __restrict__ AT,
                                                 const _Float16* __restrict__ Xin,
                                                 _Float16* __restrict__ Yout) {
  __shared__ __align__(16) _Float16 Xs[NBt * LP * Vn];  // 64 KB

  const int bc0  = blockIdx.x * NBt;
  const int wave = threadIdx.x >> 5;
  const int lane = threadIdx.x & 31;
  const int w0   = ((blockIdx.y * 8 + wave) * MBt) * 16;  // first of MBt tiles

  // async memory -> LDS fill of NBt consecutive 16x1024 f16 X-tiles
  {
    const _Float16* xin = Xin + (size_t)bc0 * LP * Vn;  // tiles are contiguous
    for (int i = threadIdx.x * 8; i < NBt * LP * Vn; i += 256 * 8) {
      const _Float16* g = xin + i;
      unsigned l = (unsigned)(uintptr_t)(Xs + i);  // flat LDS addr[31:0]
      asm volatile("global_load_async_to_lds_b128 %0, %1, off"
                   :: "v"(l), "v"(g) : "memory");
    }
    asm volatile("s_wait_asynccnt 0" ::: "memory");
  }
  __syncthreads();

  const int m  = lane & 15;   // A row (w) within tile == B col (l) within tile
  const int kg = lane >> 4;   // K group select

  const _Float16* arow  = AT + (size_t)(w0 + m) * Vn;  // tile j adds j*16*Vn
  const _Float16* bbase = Xs + m * Vn;                 // bc tile q adds q*LP*Vn

  v8f acc[MBt][NBt];
#pragma unroll
  for (int j = 0; j < MBt; ++j)
#pragma unroll
    for (int q = 0; q < NBt; ++q) acc[j][q] = (v8f){};

  // A operand: documented 16-bit A 16x32 layout (lane kg picks K {kg*8..}).
  // B operand: same layout as A (lane kg picks K {kg*8..} and {16+kg*8..}).
  auto loadA = [&](v16h (&dst)[MBt], int k0) {
#pragma unroll
    for (int j = 0; j < MBt; ++j) {
      const _Float16* aj = arow + (size_t)j * 16 * Vn + k0;
      dst[j] = ld_frag(aj + kg * 8, aj + 16 + kg * 8);
    }
  };
  auto loadB = [&](v16h (&dst)[NBt], int k0) {
#pragma unroll
    for (int q = 0; q < NBt; ++q) {
      const _Float16* bq = bbase + (size_t)q * LP * Vn + k0;
      dst[q] = ld_frag(bq + kg * 8, bq + 16 + kg * 8);
    }
  };

#define WMMA_ALL(AF, BF)                                                  \
  _Pragma("unroll")                                                       \
  for (int j = 0; j < MBt; ++j) {                                         \
    _Pragma("unroll")                                                     \
    for (int q = 0; q < NBt; ++q)                                         \
      acc[j][q] = __builtin_amdgcn_wmma_f32_16x16x32_f16(                 \
          false, AF[j], false, BF[q], (short)0, acc[j][q], false, false); \
  }

  v16h A0[MBt], B0[NBt], A1[MBt], B1[NBt];
  loadA(A0, 0);  loadB(B0, 0);
  loadA(A1, 32); loadB(B1, 32);

#pragma unroll 1
  for (int k0 = 64; k0 < Vn; k0 += 64) {
    WMMA_ALL(A0, B0);
    loadA(A0, k0); loadB(B0, k0);
    __builtin_prefetch(arow + k0 + 448, 0, 1);  // global_prefetch_b8 ahead
    WMMA_ALL(A1, B1);
    loadA(A1, k0 + 32); loadB(B1, k0 + 32);
  }
  WMMA_ALL(A0, B0);
  WMMA_ALL(A1, B1);
#undef WMMA_ALL

  // D layout: lane = (m>=8)*16 + n, vgpr i = row m&7 -> each lane stores
  // 8 consecutive w (16B contiguous) for one l, in the chainable layout.
  const int n  = lane & 15;
  const int mb = (lane >> 4) * 8;
#pragma unroll
  for (int j = 0; j < MBt; ++j)
#pragma unroll
    for (int q = 0; q < NBt; ++q) {
      v8h oh;
#pragma unroll
      for (int i = 0; i < 8; ++i) oh[i] = (_Float16)acc[j][q][i];
      *(v8h*)(Yout + ((size_t)(bc0 + q) * LP + n) * Vn + w0 + j * 16 + mb) = oh;
    }
}

// ---------------------------------------------------------------------------
// Fused channel-GEMM accumulation for one 32-channel concat piece:
//   out[b][o][v][l] (+)= sum_c piece[b*32+c][l][v] * W[o][coff+c]  (+ bias)
// grid.x = B * (V/64) = 1024, block = 256. Piece reads are v-contiguous.
// ---------------------------------------------------------------------------
__global__ __launch_bounds__(256) void acc_out(const _Float16* __restrict__ P,
                                               const float* __restrict__ W,
                                               const float* __restrict__ bias,
                                               float* __restrict__ out,
                                               int coff, int init) {
  __shared__ float Wl[COn * 32];   // 8 KB
  const int b  = blockIdx.x >> 4;
  const int v0 = (blockIdx.x & 15) * 64;
  for (int i = threadIdx.x; i < COn * 32; i += 256) {
    const int o = i >> 5, c = i & 31;
    Wl[i] = W[(size_t)o * CCAT + coff + c];
  }
  __syncthreads();
  for (int p = threadIdx.x; p < 64 * Ln; p += 256) {
    const int v = p & 63, l = p >> 6;
    float h[32];
#pragma unroll
    for (int c = 0; c < 32; ++c)
      h[c] = (float)P[(((size_t)b * Cn + c) * LP + l) * Vn + v0 + v];
    for (int o = 0; o < COn; ++o) {
      float s = 0.f;
#pragma unroll
      for (int c = 0; c < 32; ++c) s += h[c] * Wl[o * 32 + c];
      const size_t oi = (((size_t)b * COn + o) * Vn + v0 + v) * Ln + l;
      if (init) out[oi] = bias[o] + s;
      else      out[oi] += s;
    }
  }
}

// ---------------------------------------------------------------------------
extern "C" void kernel_launch(void* const* d_in, const int* in_sizes, int n_in,
                              void* d_out, int out_size, void* d_ws, size_t ws_size,
                              hipStream_t stream) {
  (void)in_sizes; (void)n_in; (void)out_size; (void)ws_size;
  const float* x    = (const float*)d_in[0];
  const float* a0   = (const float*)d_in[1];
  const float* a1   = (const float*)d_in[2];
  const float* a2   = (const float*)d_in[3];
  const float* W    = (const float*)d_in[4];
  const float* bias = (const float*)d_in[5];
  float* out = (float*)d_out;

  const size_t VV = (size_t)Vn * Vn;            // 1 Mi elements
  const size_t XSZ = (size_t)Bn * Cn * LP * Vn; // 32 Mi halves (64 MB)
  _Float16* AT = (_Float16*)d_ws;               // 3*VV halves (6 MB)
  _Float16* XT = AT + 3 * VV;
  _Float16* Y1 = XT + XSZ;
  _Float16* Y2 = Y1 + XSZ;

  const dim3 blk(256);

  // Stage 0: precision/layout conversion
  transpose_a<<<dim3(1024), blk, 0, stream>>>(a0, AT);
  transpose_a<<<dim3(1024), blk, 0, stream>>>(a1, AT + VV);
  transpose_a<<<dim3(1024), blk, 0, stream>>>(a2, AT + 2 * VV);
  pack_x<<<dim3(Bn * Cn, Vn / 64), blk, 0, stream>>>(x, XT);

  // piece 0 = x itself: initializes out with bias
  acc_out<<<dim3(Bn * (Vn / 64)), blk, 0, stream>>>(XT, W, bias, out, 0, 1);

  // pieces 1..6: two diffusion orders per adjacency, fused accumulation
  const dim3 pgrid(Bn * Cn / NBt, 64 / (8 * MBt));
  const dim3 agrid(Bn * (Vn / 64));
  const _Float16* As[3] = { AT, AT + VV, AT + 2 * VV };
  int coff = 32;
  for (int r = 0; r < 3; ++r) {
    prop_wmma<<<pgrid, blk, 0, stream>>>(As[r], XT, Y1);
    acc_out<<<agrid, blk, 0, stream>>>(Y1, W, bias, out, coff, 0); coff += 32;
    prop_wmma<<<pgrid, blk, 0, stream>>>(As[r], Y1, Y2);
    acc_out<<<agrid, blk, 0, stream>>>(Y2, W, bias, out, coff, 0); coff += 32;
  }
}
